// AttentionLayer_18992345382979
// MI455X (gfx1250) — matrix operand order repaired and checked by reading, not compiled, not run on hardware
//
#include <hip/hip_runtime.h>

// Flash-attention forward for B=8, N=2048, D=256 fp32 (Q=K=V=inputs).
// bf16 WMMA (v_wmma_f32_16x16x32_bf16) with fp32 online-softmax accumulation.
// Round 2: LDS bank-conflict padding, batched fragment loads, packed bf16 cvt.

typedef __bf16 bf16_t;
typedef __attribute__((ext_vector_type(2)))  bf16_t v2bf;
typedef __attribute__((ext_vector_type(8)))  bf16_t v8bf;
typedef __attribute__((ext_vector_type(16))) bf16_t v16bf;
typedef __attribute__((ext_vector_type(8)))  float  v8f;

#define SEQ      2048
#define DIM      256
#define DPAD     264   // DIM + 8: row stride 528B = 33 dwords*16 -> conflict-free b128
#define QTILE    64
#define KTILE    64
#define KPAD     72    // KTILE + 8: row stride 144B = 9*16
#define NWAVE    4
#define NTHREADS 128

__device__ __forceinline__ unsigned short f2bf(float f) {
  union { float f; unsigned u; } v; v.f = f;
  unsigned r = v.u + 0x7FFFu + ((v.u >> 16) & 1u);  // round-to-nearest-even
  return (unsigned short)(r >> 16);
}

// Pack two floats into two bf16 (RNE) in one dword.
__device__ __forceinline__ unsigned pk2bf(float a, float b) {
#if __has_builtin(__builtin_amdgcn_cvt_pk_bf16_f32)
  union { v2bf v; unsigned u; } c;
  c.v = __builtin_amdgcn_cvt_pk_bf16_f32(a, b);
  return c.u;
#else
  return (unsigned)f2bf(a) | ((unsigned)f2bf(b) << 16);
#endif
}

// Build a v16bf A/B fragment from two 16-byte LDS chunks.
__device__ __forceinline__ v16bf ldfrag(const unsigned short* p0, const unsigned short* p1) {
  v8bf lo = *(const v8bf*)p0;
  v8bf hi = *(const v8bf*)p1;
  return __builtin_shufflevector(lo, hi, 0,1,2,3,4,5,6,7,8,9,10,11,12,13,14,15);
}

__global__ __launch_bounds__(NTHREADS, 1)
void attn_fwd_kernel(const float* __restrict__ x, float* __restrict__ out) {
  __shared__ unsigned short Qs[QTILE][DPAD];         // pre-scaled by 1/16
  __shared__ unsigned short Ks[KTILE][DPAD];         // row-major keys
  __shared__ unsigned short VTs[DIM][KPAD];          // V transposed
  __shared__ unsigned short Ps[NWAVE][16][KPAD];     // per-wave P staging

  const int tid  = threadIdx.x;
  const int lane = tid & 31;
  const int wave = tid >> 5;
  const int m16  = lane & 15;   // row-within-frag for A, col for B/C
  const int half = lane >> 4;   // lane half selects k-group / row+8

  const int b  = blockIdx.y;
  const int q0 = blockIdx.x * QTILE;
  const float* xb = x + (size_t)b * SEQ * DIM;

  // ---- Load Q tile, fold in 1/sqrt(D)=1/16 (exact in bf16) ----
  for (int i = 0; i < (QTILE * DIM / 4) / NTHREADS; ++i) {
    int e  = i * NTHREADS + tid;
    int r  = e >> 6;            // DIM/4 = 64 float4 per row
    int c4 = e & 63;
    float4 f = ((const float4*)(xb + (size_t)(q0 + r) * DIM))[c4];
    int c = c4 * 4;
    *(unsigned*)&Qs[r][c]     = pk2bf(f.x * 0.0625f, f.y * 0.0625f);
    *(unsigned*)&Qs[r][c + 2] = pk2bf(f.z * 0.0625f, f.w * 0.0625f);
  }

  const v8f vzero = {};
  v8f acc[16];                      // O accumulator: 16 d-chunks x (16x16 f32 frag)
  float mrow[8], lrow[8];           // online-softmax stats, aligned to C layout rows
  #pragma unroll
  for (int f = 0; f < 16; ++f) acc[f] = vzero;
  #pragma unroll
  for (int v = 0; v < 8; ++v) { mrow[v] = -__builtin_inff(); lrow[v] = 0.0f; }

  for (int kt = 0; kt < SEQ / KTILE; ++kt) {
    __syncthreads();   // previous tile fully consumed
    const float* kp = xb + (size_t)kt * KTILE * DIM;
    // ---- Load K/V tile once (K == V): row-major Ks + transposed VTs ----
    for (int i = 0; i < (KTILE * DIM / 4) / NTHREADS; ++i) {
      int e  = i * NTHREADS + tid;
      int r  = e >> 6;
      int c4 = e & 63;
      float4 f = ((const float4*)(kp + (size_t)r * DIM))[c4];
      int c = c4 * 4;
      unsigned p01 = pk2bf(f.x, f.y);
      unsigned p23 = pk2bf(f.z, f.w);
      *(unsigned*)&Ks[r][c]     = p01;
      *(unsigned*)&Ks[r][c + 2] = p23;
      VTs[c + 0][r] = (unsigned short)(p01 & 0xFFFFu);
      VTs[c + 1][r] = (unsigned short)(p01 >> 16);
      VTs[c + 2][r] = (unsigned short)(p23 & 0xFFFFu);
      VTs[c + 3][r] = (unsigned short)(p23 >> 16);
    }
    if (kt + 1 < SEQ / KTILE)      // gfx1250 global_prefetch_b8 hint for next tile
      __builtin_prefetch(kp + KTILE * DIM + tid * 128, 0, 0);
    __syncthreads();

    // ---- S = (Q/16) K^T : 16x64 per wave, 8 k-chunks x 4 n-chunks ----
    v8f S[4];
    #pragma unroll
    for (int nc = 0; nc < 4; ++nc) S[nc] = vzero;
    #pragma unroll
    for (int kc = 0; kc < DIM / 32; ++kc) {
      v16bf aQ = ldfrag(&Qs[wave*16 + m16][kc*32 + half*8],
                        &Qs[wave*16 + m16][kc*32 + 16 + half*8]);
      v16bf bK[4];
      #pragma unroll
      for (int nc = 0; nc < 4; ++nc)
        bK[nc] = ldfrag(&Ks[nc*16 + m16][kc*32 + half*8],
                        &Ks[nc*16 + m16][kc*32 + 16 + half*8]);
      #pragma unroll
      for (int nc = 0; nc < 4; ++nc)
        S[nc] = __builtin_amdgcn_wmma_f32_16x16x32_bf16(false, aQ, false, bK[nc],
                                                        (short)0, S[nc], false, false);
    }

    // ---- Online softmax: row max / rescale / exp / row sum ----
    float sc[8];
    #pragma unroll
    for (int v = 0; v < 8; ++v) {
      float t = fmaxf(fmaxf(S[0][v], S[1][v]), fmaxf(S[2][v], S[3][v]));
      t = fmaxf(t, __shfl_xor(t, 1, 32));
      t = fmaxf(t, __shfl_xor(t, 2, 32));
      t = fmaxf(t, __shfl_xor(t, 4, 32));
      t = fmaxf(t, __shfl_xor(t, 8, 32));   // reduce within 16-lane half == one row
      float mnew = fmaxf(mrow[v], t);
      sc[v] = __expf(mrow[v] - mnew);
      mrow[v] = mnew;
      lrow[v] *= sc[v];
    }
    #pragma unroll
    for (int f = 0; f < 16; ++f)
      #pragma unroll
      for (int v = 0; v < 8; ++v)
        acc[f][v] *= sc[v];

    #pragma unroll
    for (int nc = 0; nc < 4; ++nc)
      #pragma unroll
      for (int v = 0; v < 8; ++v) {
        float p = __expf(S[nc][v] - mrow[v]);
        S[nc][v] = p;
        // stage P (C layout -> row-major) for the A-fragment of P*V
        Ps[wave][v + half*8][nc*16 + m16] = f2bf(p);
      }
    #pragma unroll
    for (int v = 0; v < 8; ++v) {
      float t = (S[0][v] + S[1][v]) + (S[2][v] + S[3][v]);
      t += __shfl_xor(t, 1, 32);
      t += __shfl_xor(t, 2, 32);
      t += __shfl_xor(t, 4, 32);
      t += __shfl_xor(t, 8, 32);
      lrow[v] += t;
    }

    // ---- O += P V : 2 k-chunks (keys) x 16 n-chunks (D cols), groups of 4 ----
    #pragma unroll
    for (int kb = 0; kb < KTILE / 32; ++kb) {
      v16bf aP = ldfrag(&Ps[wave][m16][kb*32 + half*8],
                        &Ps[wave][m16][kb*32 + 16 + half*8]);
      #pragma unroll
      for (int fg = 0; fg < 4; ++fg) {
        v16bf bV[4];
        #pragma unroll
        for (int j = 0; j < 4; ++j)
          bV[j] = ldfrag(&VTs[(fg*4 + j)*16 + m16][kb*32 + half*8],
                         &VTs[(fg*4 + j)*16 + m16][kb*32 + 16 + half*8]);
        #pragma unroll
        for (int j = 0; j < 4; ++j)
          acc[fg*4 + j] = __builtin_amdgcn_wmma_f32_16x16x32_bf16(
              false, aP, false, bV[j], (short)0, acc[fg*4 + j], false, false);
      }
    }
  }

  // ---- Normalize and store (C layout: row = v + 8*half, col = f*16 + m16) ----
  float inv[8];
  #pragma unroll
  for (int v = 0; v < 8; ++v) inv[v] = 1.0f / lrow[v];
  float* ob = out + ((size_t)b * SEQ + q0 + wave * 16) * DIM;
  #pragma unroll
  for (int f = 0; f < 16; ++f)
    #pragma unroll
    for (int v = 0; v < 8; ++v)
      ob[(size_t)(v + half * 8) * DIM + f * 16 + m16] = acc[f][v] * inv[v];
}

extern "C" void kernel_launch(void* const* d_in, const int* in_sizes, int n_in,
                              void* d_out, int out_size, void* d_ws, size_t ws_size,
                              hipStream_t stream) {
  const float* x = (const float*)d_in[0];
  float* out = (float*)d_out;
  dim3 grid(SEQ / QTILE, 8);   // 32 query tiles x 8 batches = 256 workgroups
  dim3 block(NTHREADS);        // 4 wave32 per workgroup
  attn_fwd_kernel<<<grid, block, 0, stream>>>(x, out);
  (void)in_sizes; (void)n_in; (void)out_size; (void)d_ws; (void)ws_size;
}
